// CharElmo_38835094290537
// MI455X (gfx1250) — matrix operand order repaired and checked by reading, not compiled, not run on hardware
//
#include <hip/hip_runtime.h>
#include <hip/hip_bf16.h>
#include <stdint.h>

// Problem constants (match reference)
#define T_LEN 256
#define B_SZ  64
#define E_DIM 512
#define H_DIM 1024
#define G4H   4096          // 4*H
#define TB    (T_LEN * B_SZ)

// ---------------- WMMA fragment types (CDNA5 wave32) ----------------
typedef __bf16 bf16x16 __attribute__((ext_vector_type(16)));
typedef float  f32x8   __attribute__((ext_vector_type(8)));

union Frag16 { bf16x16 v; uint4 q[2]; };   // 16 bf16 = 32B = 8 VGPRs
union Acc8   { f32x8 v; float f[8]; };     // 16x16 f32 C/D tile per lane

__device__ __forceinline__ unsigned short f2bf(float x) {
  unsigned u = __float_as_uint(x);
  u += 0x7FFFu + ((u >> 16) & 1u);         // round-to-nearest-even
  return (unsigned short)(u >> 16);
}
__device__ __forceinline__ float bf2f(unsigned short h) {
  return __uint_as_float(((unsigned)h) << 16);
}
__device__ __forceinline__ float sigmoidf_(float x) {
  return 1.0f / (1.0f + __expf(-x));
}

// Monotonic-counter grid barrier (all 64 WGs resident; one WG per WGP given
// the 132KB+ LDS footprint, MI455X has far more than 64 WGPs).
__device__ __forceinline__ void grid_sync(unsigned* cnt, unsigned target) {
  __syncthreads();
  if (threadIdx.x == 0) {
    __builtin_amdgcn_fence(__ATOMIC_RELEASE, "agent");
    atomicAdd(cnt, 1u);
    while (__hip_atomic_load(cnt, __ATOMIC_RELAXED, __HIP_MEMORY_SCOPE_AGENT) < target) {
      __builtin_amdgcn_s_sleep(2);
    }
  }
  __syncthreads();
  __builtin_amdgcn_fence(__ATOMIC_ACQUIRE, "agent");
}

// ---------------- tiny utility kernels ----------------
__global__ void k_init(unsigned* cnt) {
  if (threadIdx.x < 16) cnt[threadIdx.x] = 0u;
}

__global__ __launch_bounds__(256) void k_f2bf(const float* __restrict__ src,
                                              unsigned short* __restrict__ dst, int n) {
  int i = blockIdx.x * 256 + threadIdx.x;
  if (i < n) dst[i] = f2bf(src[i]);
}

__global__ void k_bias(const float* __restrict__ a, const float* __restrict__ b,
                       float* __restrict__ d) {
  int i = blockIdx.x * 256 + threadIdx.x;
  if (i < G4H) d[i] = a[i] + b[i];
}

// Embedding gather: Xbf[(t*B+b)*E + e] = bf16(embed[ids[t*B+b]][e])
__global__ __launch_bounds__(256) void k_embed(const int* __restrict__ ids,
                                               const float* __restrict__ embed,
                                               unsigned short* __restrict__ Xbf) {
  size_t idx = (size_t)blockIdx.x * 256 + threadIdx.x;   // over TB*E
  int row = (int)(idx >> 9);           // / E_DIM (512)
  int e   = (int)(idx & (E_DIM - 1));
  int tok = ids[row];
  Xbf[idx] = f2bf(embed[(size_t)tok * E_DIM + e]);
}

// ---------------- batched input-projection GEMM ----------------
// C[TB][4096] = bf16( A[TB][K] @ W[4096][K]^T + biasSum )   (f32 accumulate)
// 256 threads/block -> 8 waves. Block tile 64x64. Wave w: mtile=w&3 (rows),
// ntiles {2*(w>>2), 2*(w>>2)+1} (cols). W chunk staged in LDS (banks padded).
#define GEMM_KC        128
#define GEMM_WSTRIDE_Q 17                       // 128 data halves + 8 pad = 17 uint4
#define GEMM_LDS       (64 * GEMM_WSTRIDE_Q * 16)   // 17408 bytes

__global__ __launch_bounds__(256) void k_gemm_xg(
    const unsigned short* __restrict__ A,      // [TB][K] bf16
    const unsigned short* __restrict__ W,      // [4096][K] bf16 (row = out col)
    const float* __restrict__ biasSum,         // [4096]
    unsigned short* __restrict__ C,            // [TB][4096] bf16
    int K) {
  extern __shared__ char smem[];
  uint4* wq = (uint4*)smem;

  const int tid   = threadIdx.x;
  const int lane  = tid & 31;
  const int wave  = tid >> 5;
  const int mtile = wave & 3;
  const int npair = wave >> 2;
  const int n0    = npair * 2;
  const int hs    = lane >> 4;
  const int l15   = lane & 15;

  const int rowBlk = blockIdx.y * 64;
  const int colBlk = blockIdx.x * 64;
  const int Kq     = K >> 3;                   // uint4 per W row

  const int arow = rowBlk + mtile * 16 + l15;  // A-fragment row for this lane
  const uint4* Aq = (const uint4*)(A + (size_t)arow * K);

  Acc8 acc0, acc1;
  for (int r = 0; r < 8; ++r) { acc0.f[r] = 0.f; acc1.f[r] = 0.f; }

  for (int k0 = 0; k0 < K; k0 += GEMM_KC) {
    // stage 64 cols x 128 halves of W into LDS
    for (int i = tid; i < 64 * 16; i += 256) {
      int cl = i >> 4, qi = i & 15;
      wq[cl * GEMM_WSTRIDE_Q + qi] =
          ((const uint4*)W)[(size_t)(colBlk + cl) * Kq + (k0 >> 3) + qi];
    }
    __syncthreads();
    for (int k = 0; k < GEMM_KC; k += 32) {
      Frag16 a, b0, b1;
      int kg = k0 + k;
      // A 16x32: lane<16 holds K[0..15], lane>=16 holds K[16..31]
      a.q[0] = Aq[(kg + hs * 16) >> 3];
      a.q[1] = Aq[((kg + hs * 16) >> 3) + 1];
      // B 32x16: lane<16 = col l15 K[0..15], lane>=16 = col l15 K[16..31]
      int kb = (k + hs * 16) >> 3;
      b0.q[0] = wq[(n0 * 16 + l15) * GEMM_WSTRIDE_Q + kb];
      b0.q[1] = wq[(n0 * 16 + l15) * GEMM_WSTRIDE_Q + kb + 1];
      b1.q[0] = wq[((n0 + 1) * 16 + l15) * GEMM_WSTRIDE_Q + kb];
      b1.q[1] = wq[((n0 + 1) * 16 + l15) * GEMM_WSTRIDE_Q + kb + 1];
      acc0.v = __builtin_amdgcn_wmma_f32_16x16x32_bf16(false, a.v, false, b0.v,
                                                       (short)0, acc0.v, false, false);
      acc1.v = __builtin_amdgcn_wmma_f32_16x16x32_bf16(false, a.v, false, b1.v,
                                                       (short)0, acc1.v, false, false);
    }
    __syncthreads();
  }
  // epilogue: C-tile layout -> lane holds col l15, rows hs*8 + r
  for (int nt = 0; nt < 2; ++nt) {
    Acc8& acc = nt ? acc1 : acc0;
    int ocol = colBlk + (n0 + nt) * 16 + l15;
    float bias = biasSum[ocol];
    for (int r = 0; r < 8; ++r) {
      int orow = rowBlk + mtile * 16 + hs * 8 + r;
      C[(size_t)orow * G4H + ocol] = f2bf(acc.f[r] + bias);
    }
  }
}

// ---------------- persistent LSTM recurrence ----------------
// 64 WGs; WG `wg` owns h-columns [wg*16, wg*16+16) and its 64 Whh rows
// (4 gates x 16 cols) are pinned in LDS for all 256 steps.
#define NWG_REC       64
#define WREC_STRIDE_Q 129                       // 128 data + 1 pad uint4 per row
#define GATE_STRIDE   65
#define OFF_GATES     (64 * WREC_STRIDE_Q * 16) // 132096
#define OFF_C         (OFF_GATES + 64 * GATE_STRIDE * 4)   // +16640 = 148736
#define OFF_H         (OFF_C + 1024 * 4)                   // 152832
#define OFF_LENS      (OFF_H + 1024 * 4)                   // 156928
#define REC_LDS       (OFF_LENS + 256)                     // 157184 bytes

__global__ __launch_bounds__(256) void k_recurrent(
    const unsigned short* __restrict__ Whh,    // [4096][1024] bf16
    const unsigned short* __restrict__ Xg,     // [TB][4096] bf16 (x@Wih^T + biases)
    unsigned short* __restrict__ hbuf,         // [2][64][1024] bf16 ping-pong
    const int* __restrict__ lens,              // [64]
    float* __restrict__ out,                   // d_out base [2,T,B,2,H]
    unsigned short* __restrict__ ybf,          // [TB][1024] bf16 (layer0 only)
    unsigned* __restrict__ cnt,
    int dir, int layer, int reverse) {
  extern __shared__ char smem[];
  uint4* wq    = (uint4*)smem;
  float* gates = (float*)(smem + OFF_GATES);
  float* cST   = (float*)(smem + OFF_C);
  float* hST   = (float*)(smem + OFF_H);
  int*   lensS = (int*)(smem + OFF_LENS);

  const int tid   = threadIdx.x;
  const int wg    = blockIdx.x;
  const int lane  = tid & 31;
  const int wave  = tid >> 5;
  const int mtile = wave & 3;
  const int npair = wave >> 2;
  const int n0    = npair * 2;
  const int hs    = lane >> 4;
  const int l15   = lane & 15;

  // ---- per-layer init ----
  if (tid < 64) lensS[tid] = lens[tid];
  for (int i = tid; i < 1024; i += 256) { cST[i] = 0.f; hST[i] = 0.f; }
  for (int i = tid; i < 1024; i += 256) {            // zero own slice of h buf 0
    int b = i >> 4, j = i & 15;
    hbuf[(size_t)b * H_DIM + wg * 16 + j] = 0;
  }
  // preload Whh slice: local row r = g*16+j  ->  Whh row g*H + wg*16 + j
  for (int i = tid; i < 64 * 128; i += 256) {
    int r = i >> 7, qi = i & 127;
    int g = r >> 4, j = r & 15;
    int grow = g * H_DIM + wg * 16 + j;
    wq[r * WREC_STRIDE_Q + qi] = ((const uint4*)Whh)[(size_t)grow * 128 + qi];
  }
  unsigned phase = 1;
  grid_sync(cnt, phase * NWG_REC); phase++;

  unsigned short* hcur = hbuf;                       // read buffer
  unsigned short* hnxt = hbuf + (size_t)B_SZ * H_DIM;// write buffer
  const int arow = mtile * 16 + l15;                 // batch row for A fragment

  for (int p = 0; p < T_LEN; ++p) {
    const int t = reverse ? (T_LEN - 1 - p) : p;

    // ---- gate slice GEMM: [64 x 1024] h  @  [1024 x 64] Whh-slice ----
    const uint4* Hq = (const uint4*)(hcur + (size_t)arow * H_DIM);
    Acc8 acc0, acc1;
    for (int r = 0; r < 8; ++r) { acc0.f[r] = 0.f; acc1.f[r] = 0.f; }
    for (int k = 0; k < H_DIM; k += 32) {
      Frag16 a, b0, b1;
      a.q[0] = Hq[(k + hs * 16) >> 3];
      a.q[1] = Hq[((k + hs * 16) >> 3) + 1];
      int kb = (k + hs * 16) >> 3;
      b0.q[0] = wq[(n0 * 16 + l15) * WREC_STRIDE_Q + kb];
      b0.q[1] = wq[(n0 * 16 + l15) * WREC_STRIDE_Q + kb + 1];
      b1.q[0] = wq[((n0 + 1) * 16 + l15) * WREC_STRIDE_Q + kb];
      b1.q[1] = wq[((n0 + 1) * 16 + l15) * WREC_STRIDE_Q + kb + 1];
      acc0.v = __builtin_amdgcn_wmma_f32_16x16x32_bf16(false, a.v, false, b0.v,
                                                       (short)0, acc0.v, false, false);
      acc1.v = __builtin_amdgcn_wmma_f32_16x16x32_bf16(false, a.v, false, b1.v,
                                                       (short)0, acc1.v, false, false);
    }
    for (int nt = 0; nt < 2; ++nt) {                 // C tiles -> gates LDS
      Acc8& acc = nt ? acc1 : acc0;
      int gc = (n0 + nt) * 16 + l15;
      for (int r = 0; r < 8; ++r) {
        int grow = mtile * 16 + hs * 8 + r;
        gates[grow * GATE_STRIDE + gc] = acc.f[r];
      }
    }
    __syncthreads();

    // ---- elementwise LSTM cell update (1024 items: 64 batch x 16 cols) ----
    for (int i = tid; i < 1024; i += 256) {
      int b = i >> 4, j = i & 15;
      int hcol = wg * 16 + j;
      const size_t xb = ((size_t)t * B_SZ + b) * G4H + hcol;
      float gi = gates[b * GATE_STRIDE + j]      + bf2f(Xg[xb]);
      float gf = gates[b * GATE_STRIDE + 16 + j] + bf2f(Xg[xb + 1024]);
      float gg = gates[b * GATE_STRIDE + 32 + j] + bf2f(Xg[xb + 2048]);
      float go = gates[b * GATE_STRIDE + 48 + j] + bf2f(Xg[xb + 3072]);
      float ig = sigmoidf_(gi);
      float fg = sigmoidf_(gf);
      float g_ = tanhf(gg);
      float og = sigmoidf_(go);
      float c2 = fg * cST[i] + ig * g_;
      float h2 = og * tanhf(c2);
      bool mt  = t < lensS[b];
      float hn = mt ? h2 : hST[i];
      float cn = mt ? c2 : cST[i];
      cST[i] = cn; hST[i] = hn;
      hnxt[(size_t)b * H_DIM + hcol] = f2bf(hn);
      float y = mt ? h2 : 0.f;
      size_t ob = (((size_t)dir * T_LEN + t) * B_SZ + b) * 2;
      if (layer) y += out[(ob + 0) * H_DIM + hcol];  // residual from layer 0
      out[(ob + layer) * H_DIM + hcol] = y;
      if (ybf) ybf[((size_t)t * B_SZ + b) * H_DIM + hcol] = f2bf(y);
      int tn = reverse ? (t - 1) : (t + 1);          // prefetch next step's Xg
      if ((unsigned)tn < T_LEN)
        __builtin_prefetch(Xg + ((size_t)tn * B_SZ + b) * G4H + hcol, 0, 1);
    }

    grid_sync(cnt, phase * NWG_REC); phase++;
    { unsigned short* tmp = hcur; hcur = hnxt; hnxt = tmp; }
  }
}

// ---------------- host orchestration ----------------
extern "C" void kernel_launch(void* const* d_in, const int* in_sizes, int n_in,
                              void* d_out, int out_size, void* d_ws, size_t ws_size,
                              hipStream_t stream) {
  (void)in_sizes; (void)n_in; (void)out_size; (void)ws_size;
  const int*   ids   = (const int*)d_in[0];
  const int*   lens  = (const int*)d_in[1];
  const float* embed = (const float*)d_in[2];
  const float *Wih[4], *Whh[4], *bih[4], *bhh[4];
  int Iin[4];
  for (int s = 0; s < 4; ++s) {   // order: fw0, fw1, bw0, bw1
    int base = 3 + s * 4;
    Wih[s] = (const float*)d_in[base + 0];
    Whh[s] = (const float*)d_in[base + 1];
    bih[s] = (const float*)d_in[base + 2];
    bhh[s] = (const float*)d_in[base + 3];
    Iin[s] = (s % 2 == 0) ? E_DIM : H_DIM;
  }

  char* p = (char*)d_ws;
  auto alloc = [&](size_t bytes) {
    void* r = (void*)p;
    p += (bytes + 255) & ~(size_t)255;
    return r;
  };
  unsigned*       cnt    = (unsigned*)alloc(64);                          // 4 barriers
  unsigned short* Xbf    = (unsigned short*)alloc((size_t)TB * E_DIM * 2);// 16 MB
  unsigned short* ybf    = (unsigned short*)alloc((size_t)TB * H_DIM * 2);// 32 MB
  unsigned short* hbuf   = (unsigned short*)alloc((size_t)2 * B_SZ * H_DIM * 2);
  unsigned short* Xg     = (unsigned short*)alloc((size_t)TB * G4H * 2);  // 128 MB
  unsigned short* Wih_bf = (unsigned short*)alloc((size_t)G4H * H_DIM * 2);
  unsigned short* Whh_bf = (unsigned short*)alloc((size_t)G4H * H_DIM * 2);
  float*          biasS  = (float*)alloc(G4H * 4);
  float* out = (float*)d_out;

  k_init<<<1, 64, 0, stream>>>(cnt);
  k_embed<<<(TB * E_DIM) / 256, 256, 0, stream>>>(ids, embed, Xbf);

  for (int s = 0; s < 4; ++s) {
    int dirn = s >> 1, layer = s & 1;
    int K = Iin[s];
    k_f2bf<<<(G4H * K + 255) / 256, 256, 0, stream>>>(Wih[s], Wih_bf, G4H * K);
    k_f2bf<<<(G4H * H_DIM + 255) / 256, 256, 0, stream>>>(Whh[s], Whh_bf, G4H * H_DIM);
    k_bias<<<G4H / 256, 256, 0, stream>>>(bih[s], bhh[s], biasS);
    const unsigned short* Ain = (layer == 0) ? Xbf : ybf;
    dim3 gg(G4H / 64, TB / 64);
    k_gemm_xg<<<gg, 256, GEMM_LDS, stream>>>(Ain, Wih_bf, biasS, Xg, K);
    k_recurrent<<<NWG_REC, 256, REC_LDS, stream>>>(
        Whh_bf, Xg, hbuf, lens, out,
        (layer == 0) ? ybf : (unsigned short*)nullptr,
        cnt + s, dirn, layer, /*reverse=*/dirn);
  }
}
